// CRF_29678224015519
// MI455X (gfx1250) — compile-verified, matrix operand order checked by reading, not run on hardware
//
#include <hip/hip_runtime.h>
#include <hip/hip_bf16.h>
#include <math.h>

typedef _Float16 half_t;
typedef __attribute__((ext_vector_type(16))) _Float16 v16h;
typedef __attribute__((ext_vector_type(8)))  _Float16 v8h;
typedef __attribute__((ext_vector_type(8)))  float    v8f;

// ---------------------------------------------------------------------------
// Problem constants (from the reference)
// ---------------------------------------------------------------------------
#define BB    4
#define TT    128
#define KK    32
#define DD    512
#define HH    8
#define DH    64
#define DFFN  2048
#define NBT   (BB*TT)          // 512
#define MROWS (NBT*KK)         // 16384
#define D3    (3*DD)           // 1536

// ---------------------------------------------------------------------------
// WMMA helpers (CDNA5 wave32, v_wmma_f32_16x16x32_f16)
// ---------------------------------------------------------------------------
__device__ __forceinline__ v8f wmma16(v16h a, v16h b, v8f c) {
  return __builtin_amdgcn_wmma_f32_16x16x32_f16(false, a, false, b, (short)0, c,
                                                false, false);
}

// A-fragment (16x32 f16, M x K) loader from a row-major matrix.
// Same pattern also loads the B-fragment for B = W^T when W is row-major (N,K).
// p0 points at (row0, k0); rstride = elements between consecutive rows.
__device__ __forceinline__ v16h load_frag_a(const half_t* p0, long rstride, int lane) {
  int r    = lane & 15;
  int koff = (lane & 16) ? 8 : 0;
  const half_t* p = p0 + (long)r * rstride + koff;
  v8h lo = *(const v8h*)(p);
  v8h hi = *(const v8h*)(p + 16);
  v16h out;
#pragma unroll
  for (int j = 0; j < 8; ++j) { out[j] = lo[j]; out[8 + j] = hi[j]; }
  return out;
}

// B-fragment (32x16 f16, K x N) loader when B itself is row-major (K rows).
__device__ __forceinline__ v16h load_frag_b_km(const half_t* p0, long rstride, int lane) {
  int n    = lane & 15;
  int koff = (lane & 16) ? 8 : 0;
  v16h out;
#pragma unroll
  for (int j = 0; j < 8; ++j) {
    out[j]     = p0[(long)(koff + j) * rstride + n];
    out[8 + j] = p0[(long)(16 + koff + j) * rstride + n];
  }
  return out;
}

__device__ __forceinline__ float wave_sum(float v) {
#pragma unroll
  for (int o = 16; o > 0; o >>= 1) v += __shfl_xor(v, o, 32);
  return v;
}
__device__ __forceinline__ float wave_max(float v) {
#pragma unroll
  for (int o = 16; o > 0; o >>= 1) v = fmaxf(v, __shfl_xor(v, o, 32));
  return v;
}

// ---------------------------------------------------------------------------
// Misc small kernels
// ---------------------------------------------------------------------------
__global__ void init_kernel(int* flag) {
  if (threadIdx.x == 0 && blockIdx.x == 0) *flag = 0;
}

__global__ void cvt_kernel(const float* __restrict__ src, half_t* __restrict__ dst, long n) {
  long i = (long)blockIdx.x * blockDim.x + threadIdx.x;
  if (i < n) dst[i] = (half_t)src[i];
}

// Build the 14 concatenated features and apply W_pre (D x 14). One block per row.
__global__ __launch_bounds__(256) void pre_kernel(
    const float* __restrict__ dlen, const float* __restrict__ dir4,
    const float* __restrict__ det5, const float* __restrict__ ts2,
    const float* __restrict__ gr2,  const float* __restrict__ Wpre,
    float* __restrict__ Xf, half_t* __restrict__ Xh)
{
  long m  = blockIdx.x;            // m = bt*KK + k
  long bt = m >> 5;
  __shared__ float f[14];
  if (threadIdx.x < 14) {
    int i = threadIdx.x;
    float val;
    if      (i == 0) val = dlen[m];
    else if (i < 5)  val = dir4[m * 4 + (i - 1)];
    else if (i < 10) val = det5[m * 5 + (i - 5)];
    else if (i < 12) val = ts2[bt * 2 + (i - 10)];
    else             val = gr2[bt * 2 + (i - 12)];
    f[i] = val;
  }
  __syncthreads();
  for (int d = threadIdx.x; d < DD; d += 256) {
    float s = 0.0f;
#pragma unroll
    for (int i = 0; i < 14; ++i) s += Wpre[d * 14 + i] * f[i];
    Xf[m * DD + d] = s;
    Xh[m * DD + d] = (half_t)s;
  }
}

// ---------------------------------------------------------------------------
// Generic WMMA GEMM: Y = A(MxKd) * W(N,Kd)^T + bias ; optional relu
// One wave -> 16x64 tile of Y.  Grid must cover (M/16)*(N/64) waves exactly.
// ---------------------------------------------------------------------------
__device__ __forceinline__ void gemm_epilogue_tile(
    const v8f& c, float* outF, half_t* outH, long ldo,
    long row0, long col, const float* bias, int relu, int lane)
{
  int mbase = (lane & 16) ? 8 : 0;
  int n     = lane & 15;
#pragma unroll
  for (int v = 0; v < 8; ++v) {
    long rr = row0 + mbase + v;
    long cc = col + n;
    float x = c[v] + (bias ? bias[cc] : 0.0f);
    if (relu) x = fmaxf(x, 0.0f);
    if (outF) outF[rr * ldo + cc] = x;
    if (outH) outH[rr * ldo + cc] = (half_t)x;
  }
}

__global__ __launch_bounds__(256) void gemm_wmma_kernel(
    const half_t* __restrict__ A, int lda,
    const half_t* __restrict__ W, int ldw,
    const float* __restrict__ bias,
    float* __restrict__ outF, half_t* __restrict__ outH, int ldo,
    int M, int N, int Kd, int relu)
{
  int gw   = (int)((blockIdx.x * blockDim.x + threadIdx.x) >> 5);
  int lane = threadIdx.x & 31;
  int ntile = N >> 6;
  long row0 = (long)(gw / ntile) << 4;
  long col0 = (long)(gw % ntile) << 6;
  if (row0 >= M) return;

  v8f acc0 = {}, acc1 = {}, acc2 = {}, acc3 = {};
  for (int k0 = 0; k0 < Kd; k0 += 32) {
    const half_t* ap = A + row0 * lda + k0;
    __builtin_prefetch(ap + 32, 0, 3);   // global_prefetch of next A tile
    v16h a  = load_frag_a(ap, lda, lane);
    v16h b0 = load_frag_a(W + (col0 +  0) * ldw + k0, ldw, lane);
    v16h b1 = load_frag_a(W + (col0 + 16) * ldw + k0, ldw, lane);
    v16h b2 = load_frag_a(W + (col0 + 32) * ldw + k0, ldw, lane);
    v16h b3 = load_frag_a(W + (col0 + 48) * ldw + k0, ldw, lane);
    acc0 = wmma16(a, b0, acc0);
    acc1 = wmma16(a, b1, acc1);
    acc2 = wmma16(a, b2, acc2);
    acc3 = wmma16(a, b3, acc3);
  }
  gemm_epilogue_tile(acc0, outF, outH, ldo, row0, col0 +  0, bias, relu, lane);
  gemm_epilogue_tile(acc1, outF, outH, ldo, row0, col0 + 16, bias, relu, lane);
  gemm_epilogue_tile(acc2, outF, outH, ldo, row0, col0 + 32, bias, relu, lane);
  gemm_epilogue_tile(acc3, outF, outH, ldo, row0, col0 + 48, bias, relu, lane);
}

// ---------------------------------------------------------------------------
// Attention. xt is (K, B*T, D): seq length L=512, batch = K paths x H heads.
// One wave handles (path kp, head h, q-tile of 32 rows), flash-style streaming
// softmax over 16 key blocks of 32.  QKV buffer rows m = token*KK + kp, ld=1536.
// ---------------------------------------------------------------------------
#define ATT_WAVES 8
__global__ __launch_bounds__(256) void attn_kernel(
    const half_t* __restrict__ QKV, half_t* __restrict__ CTX)
{
  __shared__ __align__(16) float  Ssh[ATT_WAVES][32][33];
  __shared__ __align__(16) half_t Psh[ATT_WAVES][32][32];
  __shared__ float csh[ATT_WAVES][32];
  __shared__ float lsh[ATT_WAVES][32];

  int wid  = threadIdx.x >> 5;
  int lane = threadIdx.x & 31;
  int w    = blockIdx.x * ATT_WAVES + wid;
  int qt   = w & 15;
  int rem  = w >> 4;
  int h    = rem & 7;
  int kp   = rem >> 3;
  const long RS = (long)KK * D3;          // stride between consecutive tokens
  int q0 = qt * 32;

  const half_t* Qbase = QKV + (long)q0 * RS + (long)kp * D3 + h * DH;

  float mrun = -1e30f, lrun = 0.0f;
  v8f o00={},o01={},o02={},o03={},o10={},o11={},o12={},o13={};

  for (int j0 = 0; j0 < NBT; j0 += 32) {
    // ---- S = Q K^T for this 32x32 block (Kdim = 64) ----
    v8f s00={},s01={},s10={},s11={};
    const half_t* Kbase = QKV + (long)j0 * RS + (long)kp * D3 + DD + h * DH;
#pragma unroll
    for (int ks = 0; ks < DH; ks += 32) {
      v16h a0 = load_frag_a(Qbase + ks,            RS, lane);
      v16h a1 = load_frag_a(Qbase + 16 * RS + ks,  RS, lane);
      v16h b0 = load_frag_a(Kbase + ks,            RS, lane);
      v16h b1 = load_frag_a(Kbase + 16 * RS + ks,  RS, lane);
      s00 = wmma16(a0, b0, s00);  s01 = wmma16(a0, b1, s01);
      s10 = wmma16(a1, b0, s10);  s11 = wmma16(a1, b1, s11);
    }
    {
      int mb = (lane & 16) ? 8 : 0;
      int n  = lane & 15;
#pragma unroll
      for (int v = 0; v < 8; ++v) {
        Ssh[wid][mb + v][n]            = s00[v] * 0.125f;  // 1/sqrt(dh)
        Ssh[wid][mb + v][16 + n]       = s01[v] * 0.125f;
        Ssh[wid][16 + mb + v][n]       = s10[v] * 0.125f;
        Ssh[wid][16 + mb + v][16 + n]  = s11[v] * 0.125f;
      }
    }
    __syncthreads();
    // ---- online softmax: lane owns row q = lane ----
    {
      float rmax = -1e30f;
#pragma unroll
      for (int c = 0; c < 32; ++c) rmax = fmaxf(rmax, Ssh[wid][lane][c]);
      float mnew = fmaxf(mrun, rmax);
      float corr = __expf(mrun - mnew);
      float sum = 0.0f;
#pragma unroll
      for (int c = 0; c < 32; ++c) {
        float p = __expf(Ssh[wid][lane][c] - mnew);
        Psh[wid][lane][c] = (half_t)p;
        sum += p;
      }
      lrun = lrun * corr + sum;
      mrun = mnew;
      csh[wid][lane] = corr;
    }
    __syncthreads();
    // ---- rescale running ctx accumulators by per-row correction ----
    {
      int mb = (lane & 16) ? 8 : 0;
#pragma unroll
      for (int v = 0; v < 8; ++v) {
        float c0 = csh[wid][mb + v];
        float c1 = csh[wid][16 + mb + v];
        o00[v]*=c0; o01[v]*=c0; o02[v]*=c0; o03[v]*=c0;
        o10[v]*=c1; o11[v]*=c1; o12[v]*=c1; o13[v]*=c1;
      }
    }
    // ---- ctx += P(32x32) * V(32x64) ----
    v16h pa0 = load_frag_a(&Psh[wid][0][0],  32, lane);
    v16h pa1 = load_frag_a(&Psh[wid][16][0], 32, lane);
    const half_t* Vbase = QKV + (long)j0 * RS + (long)kp * D3 + 2 * DD + h * DH;
    v16h vb0 = load_frag_b_km(Vbase +  0, RS, lane);
    v16h vb1 = load_frag_b_km(Vbase + 16, RS, lane);
    v16h vb2 = load_frag_b_km(Vbase + 32, RS, lane);
    v16h vb3 = load_frag_b_km(Vbase + 48, RS, lane);
    o00 = wmma16(pa0, vb0, o00);  o01 = wmma16(pa0, vb1, o01);
    o02 = wmma16(pa0, vb2, o02);  o03 = wmma16(pa0, vb3, o03);
    o10 = wmma16(pa1, vb0, o10);  o11 = wmma16(pa1, vb1, o11);
    o12 = wmma16(pa1, vb2, o12);  o13 = wmma16(pa1, vb3, o13);
    __syncthreads();
  }

  lsh[wid][lane] = lrun;
  __syncthreads();
  {
    int mb = (lane & 16) ? 8 : 0;
    int n  = lane & 15;
#pragma unroll
    for (int v = 0; v < 8; ++v) {
      float inv0 = 1.0f / lsh[wid][mb + v];
      float inv1 = 1.0f / lsh[wid][16 + mb + v];
      long r0 = ((long)(q0 + mb + v) * KK + kp) * DD + h * DH + n;
      long r1 = ((long)(q0 + 16 + mb + v) * KK + kp) * DD + h * DH + n;
      CTX[r0 +  0] = (half_t)(o00[v] * inv0);
      CTX[r0 + 16] = (half_t)(o01[v] * inv0);
      CTX[r0 + 32] = (half_t)(o02[v] * inv0);
      CTX[r0 + 48] = (half_t)(o03[v] * inv0);
      CTX[r1 +  0] = (half_t)(o10[v] * inv1);
      CTX[r1 + 16] = (half_t)(o11[v] * inv1);
      CTX[r1 + 32] = (half_t)(o12[v] * inv1);
      CTX[r1 + 48] = (half_t)(o13[v] * inv1);
    }
  }
}

// ---------------------------------------------------------------------------
// Residual + LayerNorm over D=512. One block (128 threads) per row.
// ---------------------------------------------------------------------------
__global__ __launch_bounds__(128) void ln_kernel(
    const float* __restrict__ X, const float* __restrict__ Y,
    const float* __restrict__ g, const float* __restrict__ bb,
    float* __restrict__ outF, half_t* __restrict__ outH)
{
  long m  = blockIdx.x;
  int tid = threadIdx.x;
  __shared__ float red[128];
  const long base = m * DD;
  float v[4];
  float s = 0.0f;
#pragma unroll
  for (int i = 0; i < 4; ++i) {
    int c = tid + i * 128;
    v[i] = X[base + c] + Y[base + c];
    s += v[i];
  }
  red[tid] = s; __syncthreads();
  for (int o = 64; o > 0; o >>= 1) { if (tid < o) red[tid] += red[tid + o]; __syncthreads(); }
  float mean = red[0] / (float)DD;
  __syncthreads();
  s = 0.0f;
#pragma unroll
  for (int i = 0; i < 4; ++i) { float d = v[i] - mean; s += d * d; }
  red[tid] = s; __syncthreads();
  for (int o = 64; o > 0; o >>= 1) { if (tid < o) red[tid] += red[tid + o]; __syncthreads(); }
  float inv = rsqrtf(red[0] / (float)DD + 1e-5f);
#pragma unroll
  for (int i = 0; i < 4; ++i) {
    int c = tid + i * 128;
    float o = (v[i] - mean) * inv * g[c] + bb[c];
    outF[base + c] = o;
    if (outH) outH[base + c] = (half_t)o;
  }
}

// ---------------------------------------------------------------------------
// se = (H2 @ W_post^T) * maskf ; one wave per row
// ---------------------------------------------------------------------------
__global__ __launch_bounds__(256) void sedot_kernel(
    const float* __restrict__ H2, const float* __restrict__ Wpost,
    const unsigned char* __restrict__ mask, float* __restrict__ seraw)
{
  int gw   = (int)((blockIdx.x * blockDim.x + threadIdx.x) >> 5);
  int lane = threadIdx.x & 31;
  if (gw >= MROWS) return;
  const float* row = H2 + (long)gw * DD;
  float s = 0.0f;
  for (int c = lane; c < DD; c += 32) s += row[c] * Wpost[c];
  s = wave_sum(s);
  if (lane == 0) seraw[gw] = s * (mask[gw] ? 1.0f : 0.0f);
}

__global__ void senorm_kernel(const float* __restrict__ seraw, float* __restrict__ sen) {
  long bt = blockIdx.x;
  int k = threadIdx.x;
  float v = seraw[bt * KK + k];
  float n = sqrtf(wave_sum(v * v));
  sen[bt * KK + k] = v / fmaxf(n, 1e-12f);
}

// ---------------------------------------------------------------------------
// per-(b,t): segE, cnt, mu, sigma, abn-any (global flag)
// ---------------------------------------------------------------------------
__global__ void stats_kernel(
    const float* __restrict__ speed, const unsigned char* __restrict__ mask,
    const int* __restrict__ idx, const float* __restrict__ dist,
    const float* __restrict__ dur, float* __restrict__ segE,
    float* __restrict__ cntv, float* __restrict__ muv,
    float* __restrict__ sigv, int* __restrict__ flag)
{
  long bt = blockIdx.x;
  int k = threadIdx.x;
  long base = bt * KK;
  int  mk = mask[base + k] != 0;
  float mf = mk ? 1.0f : 0.0f;
  float sp = speed[base + k];
  float cnt = wave_sum(mf);
  int ii = idx[bt]; ii = ii < 0 ? 0 : (ii > 31 ? 31 : ii);
  float mu = speed[base + ii];
  float var = wave_sum(mf * (sp - mu) * (sp - mu)) / fmaxf(cnt, 1.0f);
  float sigma = sqrtf(fmaxf(var, 1e-30f));
  int abn = ((sp < mu - 2.0f * sigma) || (sp > mu + 2.0f * sigma)) && mk;
  float abn_any = wave_max((float)abn);
  if (k == 0) {
    cntv[bt] = cnt; muv[bt] = mu; sigv[bt] = sigma;
    int vr = mask[base] != 0;
    float d = dist[bt] / 100.0f;   // DELTA_L
    float u = dur[bt]  / 60.0f;    // DELTA_T
    float se = -(d * __logf(d)) - (u * __logf(u));
    segE[bt] = vr ? se : 0.0f;
    if (abn_any > 0.5f) atomicOr(flag, 1);
  }
}

__global__ void rect_kernel(const float* __restrict__ segE, const float* __restrict__ muv,
                            const unsigned char* __restrict__ mask, float* __restrict__ rectv)
{
  int bt = blockIdx.x * blockDim.x + threadIdx.x;
  if (bt >= NBT) return;
  int b = bt >> 7, t = bt & 127;
  auto gse = [&](int tt) -> float { return (tt < 0 || tt >= TT) ? 0.0f : segE[b * TT + tt]; };
  auto gmu = [&](int tt) -> float { return (tt < 0 || tt >= TT) ? 0.0f : muv[b * TT + tt]; };
  float den = gse(t - 2) + gse(t - 1) + gse(t + 1) + gse(t + 2);
  if (den == 0.0f) den = 1.0f;
  const float eta[2] = {0.35f, 0.15f};
  float rect = 0.0f;
#pragma unroll
  for (int j = 1; j <= 2; ++j) {
    float iv1 = gse(t - j) / den * gmu(t - j);
    float iv2 = gse(t + j) / den * gmu(t + j);
    if (iv1 != 0.0f && iv2 != 0.0f) rect += 2.0f * eta[j - 1] * (iv1 + iv2);
    else                            rect += eta[j - 1] * (gmu(t - j) + gmu(t + j));
  }
  int vr = mask[(long)bt * KK] != 0;
  rectv[bt] = vr ? rect : 0.0f;
}

__global__ void smooth_kernel(
    const float* __restrict__ speed, const unsigned char* __restrict__ mask,
    const float* __restrict__ muv, const float* __restrict__ rectv,
    const int* __restrict__ flag, float* __restrict__ smoothv, float* __restrict__ srraw)
{
  long bt = blockIdx.x;
  int k = threadIdx.x;
  long base = bt * KK;
  float sp = speed[base + k];
  float mu = muv[bt];
  int m = mask[base + k] != 0;
  float fixedv = (m && (sp != mu)) ? rectv[bt] : sp;
  float sm = (*flag) ? fixedv : sp;
  smoothv[base + k] = sm;
  float mf = m ? 1.0f : 0.0f;
  float cnt = wave_sum(mf);
  float var = wave_sum(mf * (sm - mu) * (sm - mu)) / fmaxf(cnt, 1.0f);
  if (k == 0) srraw[bt] = sqrtf(fmaxf(var, 1e-30f));
}

__global__ __launch_bounds__(128) void meanb_kernel(
    const float* __restrict__ srraw, const unsigned char* __restrict__ mask,
    float* __restrict__ srf)
{
  int b = blockIdx.x;
  int t = threadIdx.x;
  __shared__ float ssum[128];
  __shared__ float scnt[128];
  float sr = srraw[b * TT + t];
  int vr = mask[((long)(b * TT + t)) * KK] != 0;
  int nz = (sr > 0.0f) && vr;
  ssum[t] = nz ? sr : 0.0f;
  scnt[t] = nz ? 1.0f : 0.0f;
  __syncthreads();
  for (int o = 64; o > 0; o >>= 1) {
    if (t < o) { ssum[t] += ssum[t + o]; scnt[t] += scnt[t + o]; }
    __syncthreads();
  }
  float mean = ssum[0] / fmaxf(scnt[0], 1.0f);
  srf[b * TT + t] = (sr == 0.0f) ? mean : sr;
}

__device__ __forceinline__ float ncdf_diff(float z) {
  // cdf(z+2) - cdf(z-2)
  return 0.5f * (erff((z + 2.0f) * 0.70710678f) - erff((z - 2.0f) * 0.70710678f));
}

__global__ __launch_bounds__(256) void trans_kernel(
    const float* __restrict__ smoothv, const float* __restrict__ srf,
    const float* __restrict__ segE, const unsigned char* __restrict__ mask,
    float* __restrict__ tr)
{
  int bt = blockIdx.x;            // tran time slot (b*T + t)
  int t  = bt & 127;
  long obase = (long)bt * (KK * KK);
  if (t == 0) {
    for (int e = threadIdx.x; e < KK * KK; e += 256) tr[obase + e] = 0.0f;
    return;
  }
  int p = bt - 1;
  int cond = segE[bt] < segE[p];
  float sp = srf[p], sc = srf[bt];
  for (int e = threadIdx.x; e < KK * KK; e += 256) {
    int i = e >> 5, j = e & 31;
    float diff = smoothv[(long)bt * KK + j] - smoothv[(long)p * KK + i];
    float z = cond ? (diff / sp) : (-diff / sc);
    float tv = ncdf_diff(z);
    float mi = mask[(long)p * KK + i] ? 1.0f : 0.0f;
    float mj = mask[(long)bt * KK + j] ? 1.0f : 0.0f;
    tr[obase + e] = tv * mi * mj;
  }
}

__global__ void trannorm_kernel(const float* __restrict__ tr, float* __restrict__ trn) {
  long row = blockIdx.x;          // (bt*KK + i)
  int j = threadIdx.x;
  float v = tr[row * KK + j];
  float n = sqrtf(wave_sum(v * v));
  trn[row * KK + j] = v / fmaxf(n, 1e-12f);
}

// ---------------------------------------------------------------------------
// Final CRF-style forward scan + score. One wave per batch element, lane=state.
// ---------------------------------------------------------------------------
__global__ void final_kernel(const float* __restrict__ sen, const float* __restrict__ trn,
                             const unsigned char* __restrict__ mask, float* __restrict__ out)
{
  int b = blockIdx.x;
  int j = threadIdx.x;
  long bt0 = (long)b * TT;
  float s = sen[bt0 * KK + j];
  for (int t = 1; t < TT; ++t) {
    long bt = bt0 + t;
    const float* trow = trn + bt * (KK * KK);
    float sv = s;
    float mx = -1e30f;
    for (int i = 0; i < KK; ++i) {
      float si = __shfl(sv, i, 32);
      mx = fmaxf(mx, si + trow[i * KK + j]);
    }
    float sum = 0.0f;
    for (int i = 0; i < KK; ++i) {
      float si = __shfl(sv, i, 32);
      sum += __expf(si + trow[i * KK + j] - mx);
    }
    float nxt = mx + __logf(sum) + sen[bt * KK + j];
    int vr = mask[bt * KK] != 0;
    s = vr ? nxt : s;
  }
  float mx = wave_max(s);
  float sum = wave_sum(__expf(s - mx));
  float normz = mx + __logf(sum);
  if (j == 0) {
    float score = sen[bt0 * KK];
    for (int t = 1; t < TT; ++t) {
      long bt = bt0 + t;
      if (mask[bt * KK]) score += trn[bt * (KK * KK)] + sen[bt * KK];
    }
    out[b] = score - normz;
  }
}

// ---------------------------------------------------------------------------
// Host orchestration
// ---------------------------------------------------------------------------
extern "C" void kernel_launch(void* const* d_in, const int* in_sizes, int n_in,
                              void* d_out, int out_size, void* d_ws, size_t ws_size,
                              hipStream_t stream)
{
  (void)in_sizes; (void)n_in; (void)out_size; (void)ws_size;
  const float* dlen  = (const float*)d_in[0];
  const float* dir4  = (const float*)d_in[1];
  const float* det5  = (const float*)d_in[2];
  const float* ts2   = (const float*)d_in[3];
  const float* gr2   = (const float*)d_in[4];
  const float* dist  = (const float*)d_in[5];
  const float* dur   = (const float*)d_in[6];
  const float* speed = (const float*)d_in[7];
  const float* Wpre  = (const float*)d_in[8];
  const float* Wqkv  = (const float*)d_in[9];
  const float* bqkv  = (const float*)d_in[10];
  const float* Wo    = (const float*)d_in[11];
  const float* bo    = (const float*)d_in[12];
  const float* ln1g  = (const float*)d_in[13];
  const float* ln1b  = (const float*)d_in[14];
  const float* ln2g  = (const float*)d_in[15];
  const float* ln2b  = (const float*)d_in[16];
  const float* W1    = (const float*)d_in[17];
  const float* b1    = (const float*)d_in[18];
  const float* W2    = (const float*)d_in[19];
  const float* b2    = (const float*)d_in[20];
  const float* Wpost = (const float*)d_in[21];
  const unsigned char* mask = (const unsigned char*)d_in[22];
  const int* idx = (const int*)d_in[23];
  float* out = (float*)d_out;

  // ---- workspace layout (with aliasing for dead buffers) ----
  char* ws = (char*)d_ws;
  size_t off = 0;
  auto alloc = [&](size_t bytes) -> char* {
    char* p = ws + off;
    off += (bytes + 255) & ~(size_t)255;
    return p;
  };
  const long M = MROWS;
  float*  Xf    = (float*)alloc((size_t)M * DD * 4);     // later reused as Y2
  half_t* Xh    = (half_t*)alloc((size_t)M * DD * 2);
  char*   qkvr  = alloc((size_t)M * D3 * 2);
  half_t* QKVh  = (half_t*)qkvr;
  float*  Y0    = (float*)qkvr;                          // alias (QKV dead after attention)
  half_t* H1h   = (half_t*)(qkvr + (size_t)M * DD * 4);  // alias tail of QKV region
  half_t* CTXh  = (half_t*)alloc((size_t)M * DD * 2);
  float*  H1f   = (float*)alloc((size_t)M * DD * 4);
  char*   gr_   = alloc((size_t)M * DFFN * 2);
  half_t* Gh    = (half_t*)gr_;
  float*  H2f   = (float*)gr_;                           // alias (G dead after W2 gemm)
  float*  Y2f   = Xf;                                    // alias (X dead after LN1)
  half_t* Wqkv16 = (half_t*)alloc((size_t)D3 * DD * 2);
  half_t* Wo16   = (half_t*)alloc((size_t)DD * DD * 2);
  half_t* W116   = (half_t*)alloc((size_t)DFFN * DD * 2);
  half_t* W216   = (half_t*)alloc((size_t)DD * DFFN * 2);
  float*  seraw = (float*)alloc((size_t)M * 4);
  float*  sen   = (float*)alloc((size_t)M * 4);
  float*  segE  = (float*)alloc((size_t)NBT * 4);
  float*  cntv  = (float*)alloc((size_t)NBT * 4);
  float*  muv   = (float*)alloc((size_t)NBT * 4);
  float*  sigv  = (float*)alloc((size_t)NBT * 4);
  float*  rectv = (float*)alloc((size_t)NBT * 4);
  float*  srraw = (float*)alloc((size_t)NBT * 4);
  float*  srf   = (float*)alloc((size_t)NBT * 4);
  float*  smoothv = (float*)alloc((size_t)M * 4);
  int*    flag  = (int*)alloc(256);
  float*  trraw = (float*)alloc((size_t)NBT * KK * KK * 4);
  float*  trn   = (float*)alloc((size_t)NBT * KK * KK * 4);

  // 0) zero the global abn-any flag (graph-safe, re-done every call)
  init_kernel<<<1, 32, 0, stream>>>(flag);

  // 1) convert weights to f16
  auto cvt = [&](const float* s, half_t* d, long n) {
    cvt_kernel<<<(int)((n + 255) / 256), 256, 0, stream>>>(s, d, n);
  };
  cvt(Wqkv, Wqkv16, (long)D3 * DD);
  cvt(Wo,   Wo16,   (long)DD * DD);
  cvt(W1,   W116,   (long)DFFN * DD);
  cvt(W2,   W216,   (long)DD * DFFN);

  // 2) feature build + W_pre projection
  pre_kernel<<<(int)M, 256, 0, stream>>>(dlen, dir4, det5, ts2, gr2, Wpre, Xf, Xh);

  // 3) QKV = X @ Wqkv^T + bqkv  (f16 out)
  gemm_wmma_kernel<<<(M / 16) * (D3 / 64) / 8, 256, 0, stream>>>(
      Xh, DD, Wqkv16, DD, bqkv, nullptr, QKVh, D3, (int)M, D3, DD, 0);

  // 4) attention (flash-style, 32 paths x 8 heads x 16 q-tiles = 4096 waves)
  attn_kernel<<<512, 256, 0, stream>>>(QKVh, CTXh);

  // 5) Y0 = CTX @ Wo^T + bo   (f32 out, into recycled QKV region)
  gemm_wmma_kernel<<<(M / 16) * (DD / 64) / 8, 256, 0, stream>>>(
      CTXh, DD, Wo16, DD, bo, Y0, nullptr, DD, (int)M, DD, DD, 0);

  // 6) H1 = LN(X + Y0)
  ln_kernel<<<(int)M, 128, 0, stream>>>(Xf, Y0, ln1g, ln1b, H1f, H1h);

  // 7) G = relu(H1 @ W1^T + b1)  (f16 out)
  gemm_wmma_kernel<<<(M / 16) * (DFFN / 64) / 8, 256, 0, stream>>>(
      H1h, DD, W116, DD, b1, nullptr, Gh, DFFN, (int)M, DFFN, DD, 1);

  // 8) Y2 = G @ W2^T + b2  (f32 out, into recycled X region)
  gemm_wmma_kernel<<<(M / 16) * (DD / 64) / 8, 256, 0, stream>>>(
      Gh, DFFN, W216, DFFN, b2, Y2f, nullptr, DD, (int)M, DD, DFFN, 0);

  // 9) H2 = LN(H1 + Y2)
  ln_kernel<<<(int)M, 128, 0, stream>>>(H1f, Y2f, ln2g, ln2b, H2f, nullptr);

  // 10) se = (H2 @ Wpost^T)*mask ; l2norm over K
  sedot_kernel<<<(int)(M / 8), 256, 0, stream>>>(H2f, Wpost, mask, seraw);
  senorm_kernel<<<NBT, 32, 0, stream>>>(seraw, sen);

  // 11) per-(b,t) stats, rect window, smoothing, sr fixup
  stats_kernel<<<NBT, 32, 0, stream>>>(speed, mask, idx, dist, dur,
                                       segE, cntv, muv, sigv, flag);
  rect_kernel<<<2, 256, 0, stream>>>(segE, muv, mask, rectv);
  smooth_kernel<<<NBT, 32, 0, stream>>>(speed, mask, muv, rectv, flag, smoothv, srraw);
  meanb_kernel<<<BB, 128, 0, stream>>>(srraw, mask, srf);

  // 12) transitions + l2norm
  trans_kernel<<<NBT, 256, 0, stream>>>(smoothv, srf, segE, mask, trraw);
  trannorm_kernel<<<NBT * KK, 32, 0, stream>>>(trraw, trn);

  // 13) score + logsumexp forward scan
  final_kernel<<<BB, 32, 0, stream>>>(sen, trn, mask, out);
}
